// BahdanauAttention_4733053960505
// MI455X (gfx1250) — matrix operand order repaired and checked by reading, not compiled, not run on hardware
//
#include <hip/hip_runtime.h>
#include <hip/hip_bf16.h>

typedef __attribute__((ext_vector_type(16))) __bf16 v16bf;
typedef __attribute__((ext_vector_type(8)))  float  v8f;

#define B_  64
#define T_  2048
#define H_  1024
#define U_  1024
#define KT_ (H_ / 32)     // 32 k-tiles of K=32
#define UT_ (U_ / 16)     // 64 u-tiles of N=16
#define ROWS_ 64          // rows of (b,t) per block = 4 M-tiles
#define THREADS_ 512      // 16 waves
#define SMEM_BYTES_ (ROWS_ * H_ * 2 + ROWS_ * 4)   // A-tile bf16 + sscore

__device__ __forceinline__ unsigned short f32_to_bf16_rne(float f) {
    union { float f; unsigned u; } cv; cv.f = f;
    unsigned u = cv.u;
    unsigned r = u + 0x7FFFu + ((u >> 16) & 1u);   // round-to-nearest-even
    return (unsigned short)(r >> 16);
}

__device__ __forceinline__ float fast_tanh(float x) {
    // tanh(x) = 1 - 2/(exp(2x)+1) -> v_exp_f32 + v_rcp_f32 (no IEEE div expansion)
    float e = __expf(2.0f * x);
    return 1.0f - 2.0f * __builtin_amdgcn_rcpf(e + 1.0f);
}

// ---------------------------------------------------------------------------
// Kernel 0: W2 [H,U] f32 -> bf16, pre-swizzled into WMMA B-fragment order.
// Fragment (ut, kt): 32 lanes x 16 elems; lane = (u&15) + ((k&31)<16 ? 0:16),
// elem = k&15.  Flat index = ((ut*KT_+kt)*32 + lane)*16 + elem.
// ---------------------------------------------------------------------------
__global__ __launch_bounds__(256) void convert_w2_kernel(
        const float* __restrict__ W2, unsigned short* __restrict__ w2s) {
    int idx = blockIdx.x * 256 + threadIdx.x;          // idx = k*U_ + u
    int k = idx >> 10;
    int u = idx & (U_ - 1);
    unsigned short h = f32_to_bf16_rne(W2[idx]);
    int kt = k >> 5, kk = k & 31;
    int ut = u >> 4, n  = u & 15;
    int lane = n + ((kk >> 4) << 4);
    int elem = kk & 15;
    w2s[(((ut * KT_ + kt) << 5) + lane) * 16 + elem] = h;
}

// ---------------------------------------------------------------------------
// Kernel 1: qsum[b,u] = query[b,:] @ W1[:,u] + b1[u] + b2[u]   (tiny GEMM)
// ---------------------------------------------------------------------------
__global__ __launch_bounds__(256) void qsum_kernel(
        const float* __restrict__ query, const float* __restrict__ W1,
        const float* __restrict__ b1, const float* __restrict__ b2,
        float* __restrict__ qsum) {
    int b = blockIdx.x;
    int u = blockIdx.y * 256 + threadIdx.x;
    const float* q = query + b * H_;
    float acc = b1[u] + b2[u];
    #pragma unroll 4
    for (int h = 0; h < H_; ++h)
        acc += q[h] * W1[h * U_ + u];
    qsum[b * U_ + u] = acc;
}

// ---------------------------------------------------------------------------
// Kernel 2 (hot): fused  score[b,t] = tanh(qsum + values@W2) @ V + bv
// One block = 16 waves = 64 rows of (b,t)  (4 M-tiles).
// Wave w: mtile = w&3, ugroup = w>>2 -> 16 u-tiles, 32 k-steps each.
// A staged in dynamic LDS (128 KB) in WMMA A-fragment order; B streamed from
// the pre-swizzled bf16 W2 (L2-resident, shared across the 4 mtile-waves).
// ---------------------------------------------------------------------------
__global__ __launch_bounds__(THREADS_) void score_kernel(
        const float* __restrict__ values,
        const unsigned short* __restrict__ w2s,
        const float* __restrict__ qsum,
        const float* __restrict__ Vv,
        const float* __restrict__ bv,
        float* __restrict__ score) {
    extern __shared__ unsigned short smem[];
    unsigned short* ldsA = smem;                             // ROWS_*H_ bf16
    float* sscore = (float*)(smem + ROWS_ * H_);             // ROWS_ floats

    const int tid = threadIdx.x;
    const int bx  = blockIdx.x;            // B_*T_/ROWS_ = 2048 blocks
    const int b   = bx >> 5;               // T_/ROWS_ = 32 tiles per batch
    const int tt  = bx & 31;

    // ---- stage values[b, tt*64 .. +63, :] -> bf16 A-fragments in LDS
    const float* vbase = values + ((size_t)(b * T_ + tt * ROWS_)) * H_;
    #pragma unroll 4
    for (int i = 0; i < (ROWS_ * H_) / THREADS_; ++i) {
        int idx = tid + i * THREADS_;      // coalesced over h
        int m = idx >> 10;
        int k = idx & (H_ - 1);
        unsigned short h = f32_to_bf16_rne(vbase[idx]);
        int kt = k >> 5, kk = k & 31;
        // A layout (matches B packing): lanes 0-15 hold K {0..15}, lanes 16-31
        // hold K {16..31}; elem = K&15
        int hi   = kk >> 4;
        int elem = kk & 15;
        int lane = (m & 15) + (hi << 4);
        int mt   = m >> 4;
        ldsA[mt * (KT_ * 32 * 16) + (((kt << 5) + lane) << 4) + elem] = h;
    }
    if (tid < ROWS_) sscore[tid] = 0.0f;
    __syncthreads();

    const int wave   = tid >> 5;
    const int lane   = tid & 31;
    const int mtile  = wave & 3;
    const int ugroup = wave >> 2;          // 0..3, 16 u-tiles each

    const v16bf* aptr = (const v16bf*)ldsA + (mtile * KT_ * 32 + lane);
    float partial[8] = {0, 0, 0, 0, 0, 0, 0, 0};

    for (int j = 0; j < 16; ++j) {
        const int ut = ugroup * 16 + j;
        const v16bf* bptr = (const v16bf*)w2s + (ut * KT_ * 32 + lane);
        v8f acc = {};
        #pragma unroll 4
        for (int kt = 0; kt < KT_; ++kt) {
            v16bf amat = aptr[kt << 5];                     // 2x ds_load_b128
            v16bf bmat = bptr[kt << 5];                     // 2x global_load_b128
            acc = __builtin_amdgcn_wmma_f32_16x16x32_bf16(
                      false, amat, false, bmat, (short)0, acc, false, false);
        }
        // C layout: lane gives N = lane&15; VGPR r gives M = r + 8*(lane>=16)
        const int u  = ut * 16 + (lane & 15);
        const float qs = qsum[b * U_ + u];
        const float vv = Vv[u];
        #pragma unroll
        for (int r = 0; r < 8; ++r)
            partial[r] += fast_tanh(acc[r] + qs) * vv;
    }

    // reduce over the 16 N-columns inside each lane half, then merge waves
    #pragma unroll
    for (int r = 0; r < 8; ++r) {
        float v = partial[r];
        v += __shfl_xor(v, 1, 32);
        v += __shfl_xor(v, 2, 32);
        v += __shfl_xor(v, 4, 32);
        v += __shfl_xor(v, 8, 32);
        if (lane == 0)       atomicAdd(&sscore[mtile * 16 + r],     v);
        else if (lane == 16) atomicAdd(&sscore[mtile * 16 + r + 8], v);
    }
    __syncthreads();

    if (tid < ROWS_)
        score[b * T_ + tt * ROWS_ + tid] = sscore[tid] + bv[0];
}

// ---------------------------------------------------------------------------
// Kernel 3: softmax over T per batch -> attention weights (into d_out slot)
// ---------------------------------------------------------------------------
__global__ __launch_bounds__(256) void softmax_kernel(
        const float* __restrict__ score, float* __restrict__ wout) {
    __shared__ float red[256];
    const int b = blockIdx.x, tid = threadIdx.x;
    const float* s = score + b * T_;

    float loc[T_ / 256];
    float m = -3.0e38f;
    #pragma unroll
    for (int i = 0; i < T_ / 256; ++i) {
        loc[i] = s[tid + i * 256];
        m = fmaxf(m, loc[i]);
    }
    red[tid] = m; __syncthreads();
    for (int off = 128; off > 0; off >>= 1) {
        if (tid < off) red[tid] = fmaxf(red[tid], red[tid + off]);
        __syncthreads();
    }
    m = red[0]; __syncthreads();

    float sum = 0.0f;
    #pragma unroll
    for (int i = 0; i < T_ / 256; ++i) {
        loc[i] = __expf(loc[i] - m);
        sum += loc[i];
    }
    red[tid] = sum; __syncthreads();
    for (int off = 128; off > 0; off >>= 1) {
        if (tid < off) red[tid] += red[tid + off];
        __syncthreads();
    }
    const float inv = 1.0f / red[0];
    #pragma unroll
    for (int i = 0; i < T_ / 256; ++i)
        wout[b * T_ + tid + i * 256] = loc[i] * inv;
}

// ---------------------------------------------------------------------------
// Kernel 4: context[b,h] = sum_t w[b,t] * values[b,t,h]   (HBM streaming)
// ---------------------------------------------------------------------------
__global__ __launch_bounds__(256) void context_kernel(
        const float* __restrict__ values, const float* __restrict__ w,
        float* __restrict__ out) {
    const int b = blockIdx.x;
    const int h = blockIdx.y * 256 + threadIdx.x;
    const float* vb = values + (size_t)b * T_ * H_ + h;
    const float* wb = w + b * T_;
    float acc = 0.0f;
    #pragma unroll 4
    for (int t = 0; t < T_; ++t)
        acc += wb[t] * vb[(size_t)t * H_];
    out[b * H_ + h] = acc;
}

// ---------------------------------------------------------------------------
extern "C" void kernel_launch(void* const* d_in, const int* in_sizes, int n_in,
                              void* d_out, int out_size, void* d_ws, size_t ws_size,
                              hipStream_t stream) {
    const float* query  = (const float*)d_in[0];
    const float* values = (const float*)d_in[1];
    const float* W1     = (const float*)d_in[2];
    const float* b1     = (const float*)d_in[3];
    const float* W2     = (const float*)d_in[4];
    const float* b2     = (const float*)d_in[5];
    const float* Vv     = (const float*)d_in[6];
    const float* bv     = (const float*)d_in[7];

    // workspace layout
    unsigned short* w2s = (unsigned short*)d_ws;                      // 2 MB
    float* qsum  = (float*)((char*)d_ws + (size_t)2 * 1024 * 1024);   // 256 KB
    float* score = (float*)((char*)d_ws + (size_t)2 * 1024 * 1024 + 256 * 1024); // 512 KB

    float* ctx  = (float*)d_out;             // [B,H]
    float* attw = (float*)d_out + B_ * H_;   // [B,T,1]

    convert_w2_kernel<<<(H_ * U_) / 256, 256, 0, stream>>>(W2, w2s);
    qsum_kernel<<<dim3(B_, U_ / 256), 256, 0, stream>>>(query, W1, b1, b2, qsum);
    score_kernel<<<(B_ * T_) / ROWS_, THREADS_, SMEM_BYTES_, stream>>>(
        values, w2s, qsum, Vv, bv, score);
    softmax_kernel<<<B_, 256, 0, stream>>>(score, attw);
    context_kernel<<<dim3(B_, H_ / 256), 256, 0, stream>>>(values, attw, ctx);
}
